// BSGMP_61967788147123
// MI455X (gfx1250) — matrix operand order repaired and checked by reading, not compiled, not run on hardware
//
#include <hip/hip_runtime.h>
#include <hip/hip_bf16.h>

typedef __attribute__((ext_vector_type(16))) _Float16 v16h;
typedef __attribute__((ext_vector_type(8)))  _Float16 v8h;
typedef __attribute__((ext_vector_type(2)))  _Float16 v2h;
typedef __attribute__((ext_vector_type(8)))  float    v8f;
typedef __attribute__((ext_vector_type(4)))  float    v4f;

#define LDD 128          // latent dim
#define MT  4            // M-tiles (of 16 rows) per wave: B-fragment reuse x4
#define KPE 288          // padded edge-MLP input K (260 -> 288)
#define KPN 256          // node-MLP input K

__device__ __forceinline__ void wavefence() {
  // LDS ops of one wave execute in order (DScnt); this only stops the
  // compiler from reordering cross-lane LDS communication.
  __builtin_amdgcn_fence(__ATOMIC_ACQ_REL, "wavefront");
}

// ---------------- WMMA fragment loaders (CDNA5 16-bit layouts) -------------
// A 16x32 f16 (ISA 7.12.2): lane l holds row m=l&15; VGPR p holds K pair at
// kb = (p>>2)*16 + (l>>4)*8 + (p&3)*2
__device__ __forceinline__ v16h load_a_frag(const _Float16* Ain, int KP, int kc,
                                            int lmod, int lhalf) {
  v16h a;
#pragma unroll
  for (int p = 0; p < 8; ++p) {
    const int kb = ((p >> 2) << 4) + (lhalf << 3) + ((p & 3) << 1);
    v2h pa = *(const v2h*)(Ain + lmod * KP + kc + kb);
    a[2 * p] = pa.x; a[2 * p + 1] = pa.y;
  }
  return a;
}
// B 32x16 f16: lane l holds col n=l&15; VGPR p holds K pair at
// kb = (p>>2)*16 + (l>>4)*8 + (p&3)*2 (same as A)
__device__ __forceinline__ v16h load_b_frag(const _Float16* Wt, int KP, int kc,
                                            int n, int lhalf) {
  v16h b;
  const _Float16* base = Wt + (size_t)n * KP + kc + (lhalf << 3);
#pragma unroll
  for (int p = 0; p < 8; ++p) {
    v2h pb = *(const v2h*)(base + ((p >> 2) << 4) + ((p & 3) << 1));
    b[2 * p] = pb.x; b[2 * p + 1] = pb.y;
  }
  return b;
}

// MT panels of [16 x KP] (LDS,f16) @ [KP x 128] (global,f16 T) + bias -> ReLU
// -> MT panels of [16 x 128] f16 in LDS. B fragment loaded once, used MT times.
// KP is a template arg: kc-loop fully unrolled -> accumulators stay in fixed
// VGPRs (no loop-carried PHI copies, no WMMA->VALU hazard NOPs).
template <int KP>
__device__ __forceinline__ void layer_f16_mt(const _Float16* Ain, int Ast,
                                             const _Float16* __restrict__ Wt,
                                             const float* __restrict__ bias,
                                             _Float16* Aout, int Ost, int lane) {
  const int lmod = lane & 15, lhalf = lane >> 4;
  for (int t = 0; t < 8; ++t) {
    v8f acc[MT] = {};
#pragma unroll
    for (int kc = 0; kc < KP; kc += 32) {
      const v16h b = load_b_frag(Wt, KP, kc, t * 16 + lmod, lhalf);
#pragma unroll
      for (int q = 0; q < MT; ++q) {
        const v16h a = load_a_frag(Ain + q * Ast, KP, kc, lmod, lhalf);
        acc[q] = __builtin_amdgcn_wmma_f32_16x16x32_f16(false, a, false, b,
                                                        (short)0, acc[q],
                                                        false, false);
      }
    }
    const float bs = bias[t * 16 + lmod];
#pragma unroll
    for (int q = 0; q < MT; ++q)
#pragma unroll
      for (int r = 0; r < 8; ++r) {
        float v = acc[q][r] + bs;
        v = v > 0.f ? v : 0.f;
        Aout[q * Ost + (lhalf * 8 + r) * LDD + t * 16 + lmod] = (_Float16)v;
      }
  }
}

// Single-tile last layer: f32 output, no ReLU
template <int KP>
__device__ __forceinline__ void layer_f32(const _Float16* Ain,
                                          const _Float16* __restrict__ Wt,
                                          const float* __restrict__ bias,
                                          float* Obuf, int lane) {
  const int lmod = lane & 15, lhalf = lane >> 4;
  for (int t = 0; t < 8; ++t) {
    v8f acc = {};
#pragma unroll
    for (int kc = 0; kc < KP; kc += 32) {
      const v16h a = load_a_frag(Ain, KP, kc, lmod, lhalf);
      const v16h b = load_b_frag(Wt, KP, kc, t * 16 + lmod, lhalf);
      acc = __builtin_amdgcn_wmma_f32_16x16x32_f16(false, a, false, b,
                                                   (short)0, acc, false, false);
    }
    const float bs = bias[t * 16 + lmod];
#pragma unroll
    for (int r = 0; r < 8; ++r)
      Obuf[(lhalf * 8 + r) * LDD + t * 16 + lmod] = acc[r] + bs;
  }
}

// f32 row chunk (global, 16B aligned) -> packed f16 v8h stores to LDS
__device__ __forceinline__ void cvt_store64(const float* __restrict__ src,
                                            _Float16* dst) {
#pragma unroll
  for (int c8 = 0; c8 < 8; ++c8) {
    const v4f x0 = *(const v4f*)(src + c8 * 8);
    const v4f x1 = *(const v4f*)(src + c8 * 8 + 4);
    v8h o;
#pragma unroll
    for (int i = 0; i < 4; ++i) { o[i] = (_Float16)x0[i]; o[4 + i] = (_Float16)x1[i]; }
    *(v8h*)(dst + c8 * 8) = o;
  }
}

// All 32 lanes: 2 lanes per row, shfl-combine, store mean/rsqrt(var) to St.
__device__ __forceinline__ void layernorm_stats(const float* O, float* St, int lane) {
  const int row = lane >> 1, hf = lane & 1;
  const float* r = O + row * LDD + hf * 64;
  float s = 0.f, s2 = 0.f;
#pragma unroll
  for (int c4 = 0; c4 < 16; ++c4) {
    const v4f v = *(const v4f*)(r + c4 * 4);
#pragma unroll
    for (int i = 0; i < 4; ++i) { s += v[i]; s2 += v[i] * v[i]; }
  }
  s  += __shfl_xor(s, 1, 32);
  s2 += __shfl_xor(s2, 1, 32);
  if (hf == 0) {
    const float mean = s * (1.f / LDD);
    const float var  = s2 * (1.f / LDD) - mean * mean;
    St[row * 2 + 0] = mean;
    St[row * 2 + 1] = rsqrtf(var + 1e-5f);
  }
  wavefence();
}

// -------------------- Edge MLP: msg = MLP([h_i,h_j,d,|d|]); aggr[i] += msg ----
// One 32-thread block = one wave = MT tiles of 16 edges.
__global__ __launch_bounds__(32) void edge_mlp_k(
    const float* __restrict__ h, const float* __restrict__ pos,
    const int* __restrict__ g, int E,
    const _Float16* __restrict__ Wt1, const float* __restrict__ b1,
    const _Float16* __restrict__ Wt2, const float* __restrict__ b2,
    const _Float16* __restrict__ Wt3, const float* __restrict__ b3,
    const float* __restrict__ ln_s, const float* __restrict__ ln_b,
    float* __restrict__ aggr) {
  extern __shared__ char smem[];
  const int lane = threadIdx.x & 31;
  const int APAN = 16 * KPE;          // elements per A panel
  const int BPAN = 16 * LDD;
  _Float16* A  = (_Float16*)smem;                                  // MT x 16x288
  _Float16* B1 = (_Float16*)(smem + MT * APAN * 2);                // MT x 16x128
  float*    O  = (float*)(smem + MT * APAN * 2 + MT * BPAN * 2);   // 16x128 f32
  float*    St = (float*)(smem + MT * APAN * 2 + MT * BPAN * 2 + BPAN * 4);

  const int tile0 = blockIdx.x * MT;

  // gather inputs for MT tiles: 2 lanes per edge row, 64 channels per half
  for (int q = 0; q < MT; ++q) {
    const int e0 = (tile0 + q) * 16;
    const int el = lane >> 1, hf = lane & 1;
    int e = e0 + el; if (e >= E) e = E - 1;
    const int vi = g[e], vj = g[E + e];
    _Float16* Ar = A + q * APAN + el * KPE;
    cvt_store64(h + (size_t)vi * LDD + hf * 64, Ar + hf * 64);
    cvt_store64(h + (size_t)vj * LDD + hf * 64, Ar + 128 + hf * 64);
    if (lane < 16) {
      int e2 = e0 + lane; if (e2 >= E) e2 = E - 1;
      const int i2 = g[e2], j2 = g[E + e2];
      const float dx = pos[(size_t)i2 * 3 + 0] - pos[(size_t)j2 * 3 + 0];
      const float dy = pos[(size_t)i2 * 3 + 1] - pos[(size_t)j2 * 3 + 1];
      const float dz = pos[(size_t)i2 * 3 + 2] - pos[(size_t)j2 * 3 + 2];
      const float nr = sqrtf(dx * dx + dy * dy + dz * dz + 1e-12f);
      _Float16* Ag = A + q * APAN + lane * KPE;
      Ag[256] = (_Float16)dx; Ag[257] = (_Float16)dy;
      Ag[258] = (_Float16)dz; Ag[259] = (_Float16)nr;
#pragma unroll
      for (int c = 260; c < KPE; ++c) Ag[c] = (_Float16)0.f;
    }
  }
  wavefence();
  layer_f16_mt<KPE>(A, APAN, Wt1, b1, B1, BPAN, lane);
  wavefence();
  layer_f16_mt<LDD>(B1, BPAN, Wt2, b2, A, APAN, lane);  // reuse A panels
  wavefence();

  for (int q = 0; q < MT; ++q) {
    layer_f32<LDD>(A + q * APAN, Wt3, b3, O, lane);
    wavefence();
    layernorm_stats(O, St, lane);
    const int e0 = (tile0 + q) * 16;
    for (int idx = lane; idx < 16 * LDD; idx += 32) {
      const int m = idx >> 7, c = idx & 127;
      const int e = e0 + m;
      if (e >= E) continue;
      const float v = (O[idx] - St[m * 2]) * St[m * 2 + 1] * ln_s[c] + ln_b[c];
      atomicAdd(&aggr[(size_t)g[e] * LDD + c], v);
    }
    wavefence();   // O reused next iteration
  }
}

// -------------------- Node MLP: out = MLP([h,aggr]) + h (+ extra) -----------
__global__ __launch_bounds__(32) void node_mlp_k(
    const float* __restrict__ h, const float* __restrict__ aggr, int N,
    const _Float16* __restrict__ Wt1, const float* __restrict__ b1,
    const _Float16* __restrict__ Wt2, const float* __restrict__ b2,
    const _Float16* __restrict__ Wt3, const float* __restrict__ b3,
    const float* __restrict__ ln_s, const float* __restrict__ ln_b,
    const float* __restrict__ extra, float* __restrict__ out) {
  extern __shared__ char smem[];
  const int lane = threadIdx.x & 31;
  const int APAN = 16 * KPN;
  const int BPAN = 16 * LDD;
  _Float16* A  = (_Float16*)smem;                                  // MT x 16x256
  _Float16* B1 = (_Float16*)(smem + MT * APAN * 2);                // MT x 16x128
  float*    O  = (float*)(smem + MT * APAN * 2 + MT * BPAN * 2);   // 16x128 f32
  float*    St = (float*)(smem + MT * APAN * 2 + MT * BPAN * 2 + BPAN * 4);

  const int tile0 = blockIdx.x * MT;
  const int row = lane >> 1, hf = lane & 1;

  for (int q = 0; q < MT; ++q) {
    const int n0 = (tile0 + q) * 16;
    int n = n0 + row; if (n >= N) n = N - 1;
    _Float16* Ar = A + q * APAN + row * KPN;
    cvt_store64(h    + (size_t)n * LDD + hf * 64, Ar + hf * 64);
    cvt_store64(aggr + (size_t)n * LDD + hf * 64, Ar + 128 + hf * 64);
  }
  wavefence();
  layer_f16_mt<KPN>(A, APAN, Wt1, b1, B1, BPAN, lane);
  wavefence();
  layer_f16_mt<LDD>(B1, BPAN, Wt2, b2, A, APAN, lane);
  wavefence();

  for (int q = 0; q < MT; ++q) {
    layer_f32<LDD>(A + q * APAN, Wt3, b3, O, lane);
    wavefence();
    layernorm_stats(O, St, lane);
    const int n0 = (tile0 + q) * 16;
    const int n = n0 + row;
    if (n < N) {
      const float mean = St[row * 2], rs = St[row * 2 + 1];
      const float* Orow = O + row * LDD + hf * 64;
      const float* hrow = h + (size_t)n * LDD + hf * 64;
      const float* srow = ln_s + hf * 64;
      const float* brow = ln_b + hf * 64;
      float* orow = out + (size_t)n * LDD + hf * 64;
#pragma unroll
      for (int c4 = 0; c4 < 16; ++c4) {
        const v4f ov = *(const v4f*)(Orow + c4 * 4);
        const v4f sv = *(const v4f*)(srow + c4 * 4);
        const v4f bv = *(const v4f*)(brow + c4 * 4);
        const v4f hv = *(const v4f*)(hrow + c4 * 4);
        v4f res;
#pragma unroll
        for (int i = 0; i < 4; ++i)
          res[i] = (ov[i] - mean) * rs * sv[i] + bv[i] + hv[i];
        if (extra) {
          const v4f ev = *(const v4f*)(extra + (size_t)n * LDD + hf * 64 + c4 * 4);
#pragma unroll
          for (int i = 0; i < 4; ++i) res[i] += ev[i];
        }
        *(v4f*)(orow + c4 * 4) = res;
      }
    }
    wavefence();
  }
}

// -------------------- scalar / bandwidth kernels ---------------------------
__global__ void zero_k(float* p, long n) {
  for (long i = blockIdx.x * 256L + threadIdx.x; i < n; i += (long)gridDim.x * 256L)
    p[i] = 0.f;
}
// pack f32 W[K][128] -> f16 Wt[128][KP] (transposed, zero-padded rows K..KP)
__global__ void pack_k(const float* __restrict__ W, int K, int KP,
                       _Float16* __restrict__ Wt) {
  const int total = 128 * KP;
  for (int idx = blockIdx.x * 256 + threadIdx.x; idx < total; idx += gridDim.x * 256) {
    const int n = idx / KP, k = idx % KP;
    Wt[(size_t)n * KP + k] = (_Float16)(k < K ? W[(size_t)k * 128 + n] : 0.f);
  }
}
__global__ void deg_k(const int* __restrict__ g, long E, float* __restrict__ deg) {
  for (long e = blockIdx.x * 256L + threadIdx.x; e < E; e += (long)gridDim.x * 256L)
    atomicAdd(&deg[g[e]], 1.f);
}
__global__ void normw_k(const float* __restrict__ w, const float* __restrict__ deg,
                        long N, float* __restrict__ nw) {
  for (long n = blockIdx.x * 256L + threadIdx.x; n < N; n += (long)gridDim.x * 256L)
    nw[n] = w[n] / fmaxf(deg[n], 1.f);
}
__global__ void w2s_k(const int* __restrict__ g, long E, const float* __restrict__ nw,
                      float* __restrict__ w2s, float* __restrict__ aggrw) {
  for (long e = blockIdx.x * 256L + threadIdx.x; e < E; e += (long)gridDim.x * 256L) {
    const float s = nw[g[E + e]];       // normed_w[j]
    w2s[e] = s;
    atomicAdd(&aggrw[g[e]], s);         // segment over i
  }
}
__global__ void cts_k(const int* __restrict__ g, long E, const float* __restrict__ w2s,
                      const float* __restrict__ aggrw, float* __restrict__ ec) {
  for (long e = blockIdx.x * 256L + threadIdx.x; e < E; e += (long)gridDim.x * 256L)
    ec[e] = w2s[e] / fmaxf(aggrw[g[e]], 1e-12f);
}
// out[dst[e]*C+c] += ew[e]*x[src[e]*C+c]
__global__ void econv_k(const float* __restrict__ x, const int* __restrict__ dst,
                        const int* __restrict__ src, const float* __restrict__ ew,
                        long E, int C, float* __restrict__ out) {
  const long total = E * C;
  for (long idx = blockIdx.x * 256L + threadIdx.x; idx < total; idx += (long)gridDim.x * 256L) {
    const long e = idx / C; const int c = (int)(idx - e * C);
    atomicAdd(&out[(size_t)dst[e] * C + c], ew[e] * x[(size_t)src[e] * C + c]);
  }
}
__global__ void gather_k(const float* __restrict__ in, const int* __restrict__ ids,
                         long M, int C, float* __restrict__ out) {
  const long total = M * C;
  for (long idx = blockIdx.x * 256L + threadIdx.x; idx < total; idx += (long)gridDim.x * 256L) {
    const long m = idx / C; const int c = (int)(idx - m * C);
    out[(size_t)m * C + c] = in[(size_t)ids[m] * C + c];
  }
}
__global__ void scatter_k(const float* __restrict__ in, const int* __restrict__ ids,
                          long M, int C, float* __restrict__ out) {
  const long total = M * C;
  for (long idx = blockIdx.x * 256L + threadIdx.x; idx < total; idx += (long)gridDim.x * 256L) {
    const long m = idx / C; const int c = (int)(idx - m * C);
    out[(size_t)ids[m] * C + c] = in[(size_t)m * C + c];
  }
}

// ============================================================================
extern "C" void kernel_launch(void* const* d_in, const int* in_sizes, int n_in,
                              void* d_out, int out_size, void* d_ws, size_t ws_size,
                              hipStream_t stream) {
  if (n_in < 88) return;

  const float* d_h   = (const float*)d_in[0];
  const float* d_pos = (const float*)d_in[1];
  const float* d_w   = (const float*)d_in[2];
  const int*   g0    = (const int*)d_in[83];
  const int*   g1    = (const int*)d_in[84];
  const int*   g2    = (const int*)d_in[85];
  const int*   ids0  = (const int*)d_in[86];
  const int*   ids1  = (const int*)d_in[87];

  const int N0 = in_sizes[0] / LDD;
  const int N1 = in_sizes[86];
  const int N2 = in_sizes[87];
  const int E0 = in_sizes[83] / 2;
  const int E1 = in_sizes[84] / 2;
  const int E2 = in_sizes[85] / 2;

  // params tree-flatten order: bottom, down0, down1, up0, up1; each gmp =
  // edge mlp then node mlp; each mlp = W0,b0,W1,b1,W2,b2,ln_b,ln_s
  const float *W[5][2][3], *B[5][2][3], *LS[5][2], *LB[5][2];
  int pi = 3;
  for (int gq = 0; gq < 5; ++gq)
    for (int m = 0; m < 2; ++m) {
      for (int l = 0; l < 3; ++l) {
        W[gq][m][l] = (const float*)d_in[pi++];
        B[gq][m][l] = (const float*)d_in[pi++];
      }
      LB[gq][m] = (const float*)d_in[pi++];
      LS[gq][m] = (const float*)d_in[pi++];
    }

  // workspace carve
  size_t off = 0;
  auto alloc = [&](size_t bytes) -> void* {
    void* p = (char*)d_ws + off;
    off += (bytes + 255) & ~(size_t)255;
    return p;
  };
  _Float16* WT[5][2][3];
  const int KIN[2][3]  = {{260, 128, 128}, {256, 128, 128}};
  const int KPAD[2][3] = {{KPE, 128, 128}, {KPN, 128, 128}};
  for (int gq = 0; gq < 5; ++gq)
    for (int m = 0; m < 2; ++m)
      for (int l = 0; l < 3; ++l)
        WT[gq][m][l] = (_Float16*)alloc((size_t)128 * KPAD[m][l] * sizeof(_Float16));

  float* aggrbuf = (float*)alloc((size_t)N0 * LDD * 4);
  float* htmp    = (float*)alloc((size_t)N0 * LDD * 4);
  float* h0d     = (float*)alloc((size_t)N0 * LDD * 4);
  float* h0u     = (float*)alloc((size_t)N0 * LDD * 4);
  float* h1a     = (float*)alloc((size_t)N1 * LDD * 4);
  float* h1d     = (float*)alloc((size_t)N1 * LDD * 4);
  float* h2a     = (float*)alloc((size_t)N2 * LDD * 4);
  float* pos1    = (float*)alloc((size_t)N1 * 3 * 4);
  float* pos2    = (float*)alloc((size_t)N2 * 3 * 4);
  float* pbuf    = (float*)alloc((size_t)N0 * 3 * 4);
  float* deg     = (float*)alloc((size_t)N0 * 4);
  float* normw   = (float*)alloc((size_t)N0 * 4);
  float* aggrw   = (float*)alloc((size_t)N0 * 4);
  float* w2s     = (float*)alloc((size_t)E0 * 4);
  float* cts0    = (float*)alloc((size_t)E0 * 4);
  float* cts1    = (float*)alloc((size_t)E1 * 4);
  float* w1      = (float*)alloc((size_t)N1 * 4);
  (void)ws_size;

  auto nb = [](long n) -> unsigned {
    long b = (n + 255) / 256;
    if (b > 8192) b = 8192;
    if (b < 1) b = 1;
    return (unsigned)b;
  };
  auto zero = [&](float* p, long n) { zero_k<<<nb(n), 256, 0, stream>>>(p, n); };

  // pack all 30 weight matrices to transposed padded f16
  for (int gq = 0; gq < 5; ++gq)
    for (int m = 0; m < 2; ++m)
      for (int l = 0; l < 3; ++l)
        pack_k<<<nb(128L * KPAD[m][l]), 256, 0, stream>>>(W[gq][m][l], KIN[m][l],
                                                          KPAD[m][l], WT[gq][m][l]);

  // dynamic LDS per 32-thread block (MT panels + f32 out + stats), both < 64KB
  const int SME = MT * (16 * KPE) * 2 + MT * (16 * LDD) * 2 + 16 * LDD * 4 + 32 * 4;
  const int SMN = MT * (16 * KPN) * 2 + MT * (16 * LDD) * 2 + 16 * LDD * 4 + 32 * 4;

  auto run_gmp = [&](int gq, const float* hin, const float* posin, const int* gg,
                     int E, int N, const float* extra, float* out) {
    zero(aggrbuf, (long)N * LDD);
    const int etiles = (E + 15) / 16;
    edge_mlp_k<<<(etiles + MT - 1) / MT, 32, SME, stream>>>(
        hin, posin, gg, E,
        WT[gq][0][0], B[gq][0][0], WT[gq][0][1], B[gq][0][1], WT[gq][0][2], B[gq][0][2],
        LS[gq][0], LB[gq][0], aggrbuf);
    const int ntiles = (N + 15) / 16;
    node_mlp_k<<<(ntiles + MT - 1) / MT, 32, SMN, stream>>>(
        hin, aggrbuf, N,
        WT[gq][1][0], B[gq][1][0], WT[gq][1][1], B[gq][1][1], WT[gq][1][2], B[gq][1][2],
        LS[gq][1], LB[gq][1], extra, out);
  };
  auto run_calew = [&](const int* gg, int E, int N, const float* w, float* cts) {
    zero(deg, N);
    deg_k<<<nb(E), 256, 0, stream>>>(gg, E, deg);
    normw_k<<<nb(N), 256, 0, stream>>>(w, deg, N, normw);
    zero(aggrw, N);
    w2s_k<<<nb(E), 256, 0, stream>>>(gg, E, normw, w2s, aggrw);
    cts_k<<<nb(E), 256, 0, stream>>>(gg, E, w2s, aggrw, cts);
  };
  auto run_econv = [&](const float* x, const int* gg, int E, const float* ew, int C,
                       float* out, long Nout, bool aggregating) {
    zero(out, Nout * C);
    const int* dst = aggregating ? gg : gg + E;
    const int* src = aggregating ? gg + E : gg;
    econv_k<<<nb((long)E * C), 256, 0, stream>>>(x, dst, src, ew, E, C, out);
  };

  // gmp index map: 0=bottom, 1=down0, 2=down1, 3=up0, 4=up1
  // ===================== Down level 0 =====================
  run_gmp(1, d_h, d_pos, g0, E0, N0, nullptr, h0d);
  run_calew(g0, E0, N0, d_w, cts0);
  run_econv(h0d, g0, E0, cts0, LDD, htmp, N0, true);
  gather_k<<<nb((long)N1 * LDD), 256, 0, stream>>>(htmp, ids0, N1, LDD, h1a);
  run_econv(d_pos, g0, E0, cts0, 3, pbuf, N0, true);
  gather_k<<<nb((long)N1 * 3), 256, 0, stream>>>(pbuf, ids0, N1, 3, pos1);
  gather_k<<<nb((long)N1), 256, 0, stream>>>(aggrw, ids0, N1, 1, w1);

  // ===================== Down level 1 =====================
  run_gmp(2, h1a, pos1, g1, E1, N1, nullptr, h1d);
  run_calew(g1, E1, N1, w1, cts1);
  run_econv(h1d, g1, E1, cts1, LDD, htmp, N1, true);
  gather_k<<<nb((long)N2 * LDD), 256, 0, stream>>>(htmp, ids1, N2, LDD, h2a);
  run_econv(pos1, g1, E1, cts1, 3, pbuf, N1, true);
  gather_k<<<nb((long)N2 * 3), 256, 0, stream>>>(pbuf, ids1, N2, 3, pos2);

  // ===================== Bottom (level 2) =====================
  run_gmp(0, h2a, pos2, g2, E2, N2, nullptr, h2a);   // in-place (per-row residual)

  // ===================== Up to level 1 =====================
  zero(htmp, (long)N1 * LDD);
  scatter_k<<<nb((long)N2 * LDD), 256, 0, stream>>>(h2a, ids1, N2, LDD, htmp);
  run_econv(htmp, g1, E1, cts1, LDD, h1a, N1, false);  // unpool direction, reuse h1a
  run_gmp(3, h1a, pos1, g1, E1, N1, /*extra=*/h1d, h1a);

  // ===================== Up to level 0 =====================
  zero(htmp, (long)N0 * LDD);
  scatter_k<<<nb((long)N1 * LDD), 256, 0, stream>>>(h1a, ids0, N1, LDD, htmp);
  run_econv(htmp, g0, E0, cts0, LDD, h0u, N0, false);
  run_gmp(4, h0u, d_pos, g0, E0, N0, /*extra=*/h0d, (float*)d_out);
  (void)out_size;
}
